// LSTM_28741921145192
// MI455X (gfx1250) — 16-bit and 8-bit matrix operand order repaired and checked by reading, compiled, not run on hardware
//
#include <hip/hip_runtime.h>
#include <cstdint>
#include <cstddef>


typedef __attribute__((ext_vector_type(16))) _Float16     v16h;
typedef __attribute__((ext_vector_type(8)))  _Float16     v8h;
typedef __attribute__((ext_vector_type(8)))  float        v8f;
typedef __attribute__((ext_vector_type(8)))  int          v8i;
typedef __attribute__((ext_vector_type(4)))  int          v4i;
typedef __attribute__((ext_vector_type(4)))  unsigned int u32x4;

#define HS    256
#define G4    1024
#define SEQ   2048
#define BATCH 64
#define USE_TDM 1

static constexpr size_t XU_BYTES = (size_t)SEQ * BATCH * G4 * 2;
static constexpr size_t WS_B     = XU_BYTES;
static constexpr size_t WS_U     = WS_B + 4096;
static constexpr size_t WS_V     = WS_U + 524288;

static constexpr unsigned LDS_H    = 0;
static constexpr unsigned LDS_XU   = 8704;
static constexpr unsigned LDS_SIZE = 8704 + 2 * 32768;


__device__ __forceinline__ float sigmoidf_(float x) {
    return 1.0f / (1.0f + __expf(-x));
}
__device__ __forceinline__ float tanhf_(float x) {
    return 2.0f / (1.0f + __expf(-2.0f * x)) - 1.0f;
}

__device__ __forceinline__ unsigned char f32_to_e4m3(float xf) {
    unsigned u = __float_as_uint(xf);
    unsigned char s = (unsigned char)((u >> 24) & 0x80u);
    unsigned ax = u & 0x7fffffffu;
    if (ax >= 0x7f800000u) return (unsigned char)(s | 0x7f);
    if (ax >= 0x43e00000u) return (unsigned char)(s | 0x7e);
    int e = (int)(ax >> 23) - 127;
    if (e < -6) {
        float q = __uint_as_float(ax) * 512.0f;
        int m = (int)(q + 0.5f);
        if (m >= 8) return (unsigned char)(s | 0x08);
        return (unsigned char)(s | (unsigned char)m);
    }
    unsigned mant = ax & 0x7fffffu;
    unsigned r = mant + 0x80000u;
    if (r & 0x800000u) { e += 1; r = 0; if (e > 8) return (unsigned char)(s | 0x7e); }
    unsigned m3 = (r >> 20) & 7u;
    return (unsigned char)(s | (unsigned char)(((e + 7) << 3) | m3));
}

#if USE_TDM
__device__ __forceinline__ void tdm_load_xu(unsigned lds_off, unsigned long long gaddr) {
    u32x4 g0;
    g0[0] = 1u;
    g0[1] = lds_off;
    g0[2] = (unsigned)gaddr;
    g0[3] = (unsigned)((gaddr >> 32) & 0x1ffffffu) | (2u << 30);
    v8i g1;
    g1[0] = 0x00020000;
    g1[1] = (int)(128u << 16);
    g1[2] = (int)(8u << 16);
    g1[3] = (int)(128u << 16);
    g1[4] = 8;
    g1[5] = 1024;
    g1[6] = 0;
    g1[7] = 0;
    v4i z4 = {0, 0, 0, 0};
#if __clang_major__ >= 23
    v8i z8 = {0, 0, 0, 0, 0, 0, 0, 0};
    __builtin_amdgcn_tensor_load_to_lds(g0, g1, z4, z4, z8, 0);
#else
    __builtin_amdgcn_tensor_load_to_lds(g0, g1, z4, z4, 0);
#endif
}
#endif


__global__ void pack_u_f16(const float* __restrict__ Ui, const float* __restrict__ Uf,
                           const float* __restrict__ Uc, const float* __restrict__ Uo,
                           _Float16* __restrict__ wsU) {
    int idx = blockIdx.x * 256 + threadIdx.x;
    int bid = idx >> 9;
    int w   = idx & 511;
    int nt = bid >> 3, kc = bid & 7;
    int l = w >> 4, r = w & 15;
    int v = r >> 1, hh = r & 1;
    int K = kc * 32 + ((r & 8) ? 16 : 0) + ((l & 16) ? 8 : 0) + (r & 7);
    int N = nt * 16 + (l & 15);
    const float* Ug = (N < 256) ? Ui : (N < 512) ? Uf : (N < 768) ? Uc : Uo;
    wsU[idx] = (_Float16)Ug[K * HS + (N & 255)];
}

__global__ void pack_v_fp8(const float* __restrict__ Vi, const float* __restrict__ Vf,
                           const float* __restrict__ Vc, const float* __restrict__ Vo,
                           unsigned char* __restrict__ wsV) {
    int idx = blockIdx.x * 256 + threadIdx.x;
    int bid = idx >> 10;
    int w   = idx & 1023;
    int nt = bid >> 2, kc = bid & 3;
    int l = w >> 5, p = w & 31;
    int v = p >> 2, bb = p & 3;
    int Kl = ((v & 4) ? 32 : 0) + ((l & 16) ? 16 : 0) + (v & 3) * 4 + bb;
    int K = kc * 64 + Kl;
    int N = nt * 16 + (l & 15);
    const float* Vg = (N < 256) ? Vi : (N < 512) ? Vf : (N < 768) ? Vc : Vo;
    wsV[idx] = f32_to_e4m3(Vg[K * HS + (N & 255)]);
}

__global__ void pack_bias(const float* __restrict__ bi, const float* __restrict__ bf,
                          const float* __restrict__ bc, const float* __restrict__ bo,
                          float* __restrict__ wsb) {
    int i = blockIdx.x * 256 + threadIdx.x;
    const float* bg = (i < 256) ? bi : (i < 512) ? bf : (i < 768) ? bc : bo;
    wsb[i] = bg[i & 255];
}

__global__ __launch_bounds__(256) void xu_gemm_f16(
        const float* __restrict__ x, const _Float16* __restrict__ wsU,
        const float* __restrict__ wsb, _Float16* __restrict__ xu) {
    const int lane = threadIdx.x & 31;
    const int wave = threadIdx.x >> 5;
    const int waveIdx = blockIdx.x * 8 + wave;
    const int wg = waveIdx >> 9;
    const int s0 = (waveIdx & 511) * 4;

    const int arow = lane & 15;
    const int klo  = (lane & 16) ? 8 : 0;
    v16h A[4][8];
#pragma unroll
    for (int mi = 0; mi < 4; ++mi) {
#pragma unroll
        for (int kc = 0; kc < 8; ++kc) {
            const float* xp = x + ((size_t)(wg * 16 + arow) * SEQ + (s0 + mi)) * HS
                              + kc * 32 + klo;
            float4 c0a = *(const float4*)(xp);
            float4 c0b = *(const float4*)(xp + 4);
            float4 c1a = *(const float4*)(xp + 16);
            float4 c1b = *(const float4*)(xp + 20);
            v16h a;
            a[0]=(_Float16)c0a.x; a[1]=(_Float16)c0a.y; a[2]=(_Float16)c0a.z; a[3]=(_Float16)c0a.w;
            a[4]=(_Float16)c0b.x; a[5]=(_Float16)c0b.y; a[6]=(_Float16)c0b.z; a[7]=(_Float16)c0b.w;
            a[8]=(_Float16)c1a.x; a[9]=(_Float16)c1a.y; a[10]=(_Float16)c1a.z; a[11]=(_Float16)c1a.w;
            a[12]=(_Float16)c1b.x; a[13]=(_Float16)c1b.y; a[14]=(_Float16)c1b.z; a[15]=(_Float16)c1b.w;
            A[mi][kc] = a;
        }
    }

    const int ncol = lane & 15;
#pragma unroll 1
    for (int nt = 0; nt < 64; ++nt) {
        float bv = wsb[nt * 16 + ncol];
        v8f acc[4];
#pragma unroll
        for (int mi = 0; mi < 4; ++mi)
#pragma unroll
            for (int v = 0; v < 8; ++v) acc[mi][v] = bv;
#pragma unroll
        for (int kc = 0; kc < 8; ++kc) {
            v16h b = *(const v16h*)(wsU + (size_t)(nt * 8 + kc) * 512 + lane * 16);
#pragma unroll
            for (int mi = 0; mi < 4; ++mi)
                acc[mi] = __builtin_amdgcn_wmma_f32_16x16x32_f16(
                    false, A[mi][kc], false, b, (short)0, acc[mi], false, false);
        }
#pragma unroll
        for (int mi = 0; mi < 4; ++mi) {
            v8h hv;
#pragma unroll
            for (int v = 0; v < 8; ++v) hv[v] = (_Float16)acc[mi][v];
            size_t off = (((size_t)(s0 + mi) * 4 + wg) * 64 + nt) * 32 + lane;
            ((v8h*)xu)[off] = hv;
        }
    }
}

__global__ __launch_bounds__(256) void lstm_scan(
        const _Float16* __restrict__ xu, const unsigned char* __restrict__ wsV,
        float* __restrict__ out) {
    extern __shared__ __align__(16) char smem[];
    unsigned char* hB = (unsigned char*)(smem + LDS_H);

    const int lane = threadIdx.x & 31;
    const int wv   = __builtin_amdgcn_readfirstlane((int)(threadIdx.x >> 5));
    const int wg   = blockIdx.x;

    for (int i = threadIdx.x; i < 2 * 16 * 272; i += 256) hB[i] = 0;

    v8i Bf[8][4];
#pragma unroll
    for (int nt = 0; nt < 8; ++nt) {
        int t = wv + nt * 8;
#pragma unroll
        for (int kc = 0; kc < 4; ++kc)
            Bf[nt][kc] = *(const v8i*)(wsV + (size_t)(t * 4 + kc) * 1024 + lane * 32);
    }

    float c[2][8];
#pragma unroll
    for (int g = 0; g < 2; ++g)
#pragma unroll
        for (int v = 0; v < 8; ++v) c[g][v] = 0.f;

    const int ncol  = lane & 15;
    const int mbase = (lane >> 4) * 8;
    const int hilo  = (lane & 16) ? 8 : 0;

    const unsigned long long gbase = (unsigned long long)(uintptr_t)xu;
#if USE_TDM
    tdm_load_xu(LDS_XU + (unsigned)wv * 4096u,
                gbase + (unsigned long long)((0 * 4 + wg) * 32768 + wv * 512));
#endif
    __syncthreads();

    for (int s = 0; s < SEQ; ++s) {
        const int par = s & 1;
        const unsigned char* hb = hB + par * 4352;
        unsigned char*       hn = hB + (par ^ 1) * 4352;

#if USE_TDM
        if (s + 1 < SEQ) {
            tdm_load_xu(LDS_XU + (unsigned)((par ^ 1) * 32768 + wv * 4096),
                        gbase + (unsigned long long)(((s + 1) * 4 + wg) * 32768 + wv * 512));
            __builtin_amdgcn_s_wait_tensorcnt(1);
        } else {
            __builtin_amdgcn_s_wait_tensorcnt(0);
        }
#endif

        v8i Af[4];
#pragma unroll
        for (int kc = 0; kc < 4; ++kc) {
            v8i a;
#pragma unroll
            for (int pp = 0; pp < 4; ++pp) {
                unsigned long long q = *(const unsigned long long*)
                    (hb + (lane & 15) * 272 + kc * 64 + (pp >> 1) * 32 + (pp & 1) * 8 + 2 * hilo);
                a[2 * pp]     = (int)(unsigned)q;
                a[2 * pp + 1] = (int)(unsigned)(q >> 32);
            }
            Af[kc] = a;
        }

        v8f z[8];
#pragma unroll
        for (int nt = 0; nt < 8; ++nt) {
            v8h hx;
#if USE_TDM
            hx = *(const v8h*)(smem + LDS_XU + par * 32768 + wv * 4096
                              + nt * 512 + lane * 16);
#else
            hx = ((const v8h*)xu)[(((size_t)s * 4 + wg) * 64 + (wv + nt * 8)) * 32 + lane];
#endif
            v8f acc;
#pragma unroll
            for (int v = 0; v < 8; ++v) acc[v] = (float)hx[v];
#pragma unroll
            for (int kc = 0; kc < 4; ++kc)
                acc = __builtin_amdgcn_wmma_f32_16x16x64_fp8_fp8(
                    Af[kc], Bf[nt][kc], (short)0, acc, false, false);
            z[nt] = acc;
        }

#pragma unroll
        for (int g = 0; g < 2; ++g) {
            int jcol = wv * 16 + g * 128 + ncol;
#pragma unroll
            for (int v = 0; v < 8; ++v) {
                float iv = sigmoidf_(z[0 + g][v]);
                float fv = sigmoidf_(z[2 + g][v]);
                float gv = tanhf_(z[4 + g][v]);
                float ov = sigmoidf_(z[6 + g][v]);
                float cn = fv * c[g][v] + iv * gv;
                c[g][v] = cn;
                float hv = ov * tanhf_(cn);
                int M = mbase + v;
                hn[M * 272 + jcol] = f32_to_e4m3(hv);
                if (s == SEQ - 1) {
                    int brow = wg * 16 + M;
                    out[brow * HS + jcol] = hv;
                    out[BATCH * HS + brow * HS + jcol] = cn;
                }
            }
        }
        __syncthreads();
    }
}


extern "C" void kernel_launch(void* const* d_in, const int* in_sizes, int n_in,
                              void* d_out, int out_size, void* d_ws, size_t ws_size,
                              hipStream_t stream) {
    (void)in_sizes; (void)n_in; (void)out_size; (void)ws_size;
    const float* x  = (const float*)d_in[0];
    const float* Ui = (const float*)d_in[1];
    const float* Vi = (const float*)d_in[2];
    const float* bi = (const float*)d_in[3];
    const float* Uf = (const float*)d_in[4];
    const float* Vf = (const float*)d_in[5];
    const float* bf = (const float*)d_in[6];
    const float* Uc = (const float*)d_in[7];
    const float* Vc = (const float*)d_in[8];
    const float* bc = (const float*)d_in[9];
    const float* Uo = (const float*)d_in[10];
    const float* Vo = (const float*)d_in[11];
    const float* bo = (const float*)d_in[12];

    char* ws = (char*)d_ws;
    _Float16*      xu  = (_Float16*)(ws);
    float*         wsb = (float*)(ws + WS_B);
    _Float16*      wsU = (_Float16*)(ws + WS_U);
    unsigned char* wsV = (unsigned char*)(ws + WS_V);

    pack_u_f16<<<1024, 256, 0, stream>>>(Ui, Uf, Uc, Uo, wsU);
    pack_v_fp8<<<1024, 256, 0, stream>>>(Vi, Vf, Vc, Vo, wsV);
    pack_bias <<<4,    256, 0, stream>>>(bi, bf, bc, bo, wsb);
    xu_gemm_f16<<<256, 256, 0, stream>>>(x, wsU, wsb, xu);
    lstm_scan  <<<4,   256, LDS_SIZE, stream>>>(xu, wsV, (float*)d_out);
}
